// MultiHeadSelfAttention_70557722738735
// MI455X (gfx1250) — matrix operand order repaired and checked by reading, not compiled, not run on hardware
//
#include <hip/hip_runtime.h>

typedef _Float16 v16h __attribute__((ext_vector_type(16)));
typedef _Float16 h8   __attribute__((ext_vector_type(8)));
typedef float    v8f  __attribute__((ext_vector_type(8)));

#define D_MODEL 1024
#define N_HEADS 16
#define D_HEAD  64
#define N_CTX   2048
#define BATCH   2
#define ROWS    (BATCH * N_CTX)   // 4096

__device__ __forceinline__ v8f wmma16(v16h a, v16h b, v8f c) {
  // D = A(16x32 f16) * B(32x16 f16) + C(16x16 f32)
  return __builtin_amdgcn_wmma_f32_16x16x32_f16(false, a, false, b, (short)0, c,
                                                false, false);
}

__device__ __forceinline__ v16h ld_frag(const _Float16* p0, const _Float16* p1) {
  h8 a = *(const h8*)p0;
  h8 b = *(const h8*)p1;
  v16h r;
#pragma unroll
  for (int i = 0; i < 8; ++i) { r[i] = a[i]; r[8 + i] = b[i]; }
  return r;
}

// ---------------------------------------------------------------------------
// Weight transpose + f32->f16: Wt[n][k] = (f16)W[k][n].  LDS-tiled 32x32.
// ---------------------------------------------------------------------------
__global__ __launch_bounds__(256) void wt_kernel(const float* __restrict__ W,
                                                 _Float16* __restrict__ Wt) {
  __shared__ _Float16 t[32][33];
  const int tx = threadIdx.x, ty = threadIdx.y;           // 32 x 8
  const int n0 = blockIdx.x * 32, k0 = blockIdx.y * 32;
#pragma unroll
  for (int i = 0; i < 4; ++i)
    t[ty + 8 * i][tx] = (_Float16)W[(size_t)(k0 + ty + 8 * i) * D_MODEL + n0 + tx];
  __syncthreads();
#pragma unroll
  for (int i = 0; i < 4; ++i)
    Wt[(size_t)(n0 + ty + 8 * i) * D_MODEL + k0 + tx] = t[tx][ty + 8 * i];
}

// ---------------------------------------------------------------------------
// LayerNorm: one block per row of 1024; writes f16 Xn.
// ---------------------------------------------------------------------------
__global__ __launch_bounds__(256) void ln_kernel(const float* __restrict__ X,
                                                 const float* __restrict__ w,
                                                 const float* __restrict__ bn,
                                                 _Float16* __restrict__ Xn) {
  __shared__ float red[256];
  const int row = blockIdx.x;
  const int tid = threadIdx.x;
  const float* xr = X + (size_t)row * D_MODEL;

  float x[4];
  float s = 0.f;
#pragma unroll
  for (int i = 0; i < 4; ++i) { x[i] = xr[tid + 256 * i]; s += x[i]; }
  red[tid] = s; __syncthreads();
  for (int off = 128; off > 0; off >>= 1) {
    if (tid < off) red[tid] += red[tid + off];
    __syncthreads();
  }
  const float mu = red[0] * (1.0f / D_MODEL);
  __syncthreads();

  float vs = 0.f;
#pragma unroll
  for (int i = 0; i < 4; ++i) { float d = x[i] - mu; vs += d * d; }
  red[tid] = vs; __syncthreads();
  for (int off = 128; off > 0; off >>= 1) {
    if (tid < off) red[tid] += red[tid + off];
    __syncthreads();
  }
  const float rstd = rsqrtf(red[0] * (1.0f / D_MODEL) + 1e-5f);

#pragma unroll
  for (int i = 0; i < 4; ++i) {
    const int c = tid + 256 * i;
    Xn[(size_t)row * D_MODEL + c] = (_Float16)((x[i] - mu) * rstd * w[c] + bn[c]);
  }
}

// ---------------------------------------------------------------------------
// WMMA GEMM: C[4096x1024] = scale * (A(f16) @ Wt^T + bias); Wt is f16 [N][K].
// Each wave owns a 32x64 strip: acc[2][4]; per 32-wide k-step the 4 B-frags
// are reused across 2 A-frags -> 8 wmma / 12 b128 loads.
// MODE 0: f16 head layout [B, H, S, 64]         (Q with scale=1/8, K)
// MODE 2: f16 transposed head layout [B,H,64,S] (V)
// MODE 1: f32 d_out [rows, 1024]                (final projection)
// ---------------------------------------------------------------------------
template <int MODE>
__global__ __launch_bounds__(256) void gemm16(const _Float16* __restrict__ A,
                                              const _Float16* __restrict__ Wt,
                                              const float* __restrict__ bias,
                                              void* __restrict__ dst, float scale) {
  const int wid  = blockIdx.x * (blockDim.x >> 5) + (threadIdx.x >> 5); // 0..2047
  const int lane = threadIdx.x & 31;
  const int l16  = lane & 15;
  const int half = lane >> 4;
  const int tm = wid >> 4;   // 0..127  (32-row tile)
  const int tn = wid & 15;   // 0..15   (64-col strip)

  const _Float16* arow0 = A + (size_t)(tm * 32 + l16) * D_MODEL;
  const _Float16* arow1 = arow0 + (size_t)16 * D_MODEL;
  v8f acc[2][4] = {};

  for (int k0 = 0; k0 < D_MODEL; k0 += 32) {
    // A fragments: k = k0 + (e/8)*16 + 8*half + e%8 (two contiguous h8 runs)
    v16h a0 = ld_frag(arow0 + k0 + 8 * half, arow0 + k0 + 16 + 8 * half);
    v16h a1 = ld_frag(arow1 + k0 + 8 * half, arow1 + k0 + 16 + 8 * half);
#pragma unroll
    for (int nt = 0; nt < 4; ++nt) {
      const int n = tn * 64 + nt * 16 + l16;
      // B fragment: k = k0 + (e/8)*16 + 8*half + e%8 (two h8 runs in transposed weights)
      const _Float16* wp = Wt + (size_t)n * D_MODEL + k0 + 8 * half;
      v16h bm = ld_frag(wp, wp + 16);
      acc[0][nt] = wmma16(a0, bm, acc[0][nt]);
      acc[1][nt] = wmma16(a1, bm, acc[1][nt]);
    }
  }

#pragma unroll
  for (int ms = 0; ms < 2; ++ms)
#pragma unroll
    for (int nt = 0; nt < 4; ++nt) {
      const int n = tn * 64 + nt * 16 + l16;
      const float bv = bias[n];
#pragma unroll
      for (int r = 0; r < 8; ++r) {
        const int orow = tm * 32 + ms * 16 + r + 8 * half;  // global row b*S+s
        const float val = (acc[ms][nt][r] + bv) * scale;
        if (MODE == 1) {
          ((float*)dst)[(size_t)orow * D_MODEL + n] = val;
        } else {
          const int h = n >> 6, dh = n & 63;
          const int b = orow >> 11, s = orow & 2047;
          if (MODE == 0) {
            ((_Float16*)dst)[(((size_t)(b * N_HEADS + h) * N_CTX) + s) * D_HEAD + dh] =
                (_Float16)val;
          } else {  // MODE == 2
            ((_Float16*)dst)[(((size_t)(b * N_HEADS + h) * D_HEAD) + dh) * N_CTX + s] =
                (_Float16)val;
          }
        }
      }
    }
}

// ---------------------------------------------------------------------------
// One 32-key flash-attention chunk.  MASKED instantiation only runs for the
// 1-2 diagonal chunks; the steady-state path has no compares/selects.
// Scores are already scaled (1/sqrt(Dh) folded into Q projection).
// ---------------------------------------------------------------------------
template <bool MASKED>
__device__ __forceinline__ void attn_chunk(int kk0, int qg, int l16, int half,
                                           const _Float16* __restrict__ Kp,
                                           const _Float16* __restrict__ Vp,
                                           const v16h (&qb)[2],
                                           float& mrun, float& lrun, v8f (&O)[4]) {
  // --- scores^T for two 16-key subtiles: c[r] = s[query l16][key r+8*half]
  v8f sc[2];
#pragma unroll
  for (int st = 0; st < 2; ++st) {
    v8f a = {};
#pragma unroll
    for (int kh = 0; kh < 2; ++kh) {
      const _Float16* kp = Kp + (size_t)(kk0 + st * 16 + l16) * D_HEAD + kh * 32;
      v16h ka = ld_frag(kp + 8 * half, kp + 16 + 8 * half);
      a = wmma16(ka, qb[kh], a);
    }
    sc[st] = a;
  }

  // --- (mask +) chunk row-max
  float cmax = -1e30f;
#pragma unroll
  for (int st = 0; st < 2; ++st)
#pragma unroll
    for (int r = 0; r < 8; ++r) {
      float v = sc[st][r];
      if (MASKED) {
        const int key = kk0 + st * 16 + r + 8 * half;
        v = (key <= qg) ? v : -1e30f;
        sc[st][r] = v;
      }
      cmax = fmaxf(cmax, v);
    }
  cmax = fmaxf(cmax, __shfl_xor(cmax, 16));

  const float mnew  = fmaxf(mrun, cmax);
  const float alpha = __expf(mrun - mnew);
  float rs = 0.f;
  v16h pa;   // AV A-fragment: pa[e] = p[e/8][e%8] (same-lane mapping)
#pragma unroll
  for (int st = 0; st < 2; ++st)
#pragma unroll
    for (int r = 0; r < 8; ++r) {
      const float pv = __expf(sc[st][r] - mnew);
      rs += pv;
      pa[st * 8 + r] = (_Float16)pv;
    }
  rs += __shfl_xor(rs, 16);
  lrun = lrun * alpha + rs;
  mrun = mnew;

  // --- rescale O rows (row of c[r] is query r+8*half -> broadcast alpha)
  float ar[8];
#pragma unroll
  for (int r = 0; r < 8; ++r) ar[r] = __shfl(alpha, r + 8 * half);

#pragma unroll
  for (int nt = 0; nt < 4; ++nt) {
    // V B-fragment from transposed V: k = (e/8)*16 + 8*half + e%8 per lane
    const _Float16* vp = Vp + (size_t)(nt * 16 + l16) * N_CTX + kk0 + 8 * half;
    v16h vb = ld_frag(vp, vp + 16);
#pragma unroll
    for (int r = 0; r < 8; ++r) O[nt][r] *= ar[r];
    O[nt] = wmma16(pa, vb, O[nt]);
  }
}

// ---------------------------------------------------------------------------
// Flash attention, one wave per (b,h,16-query tile).
// Scores computed TRANSPOSED (S^T = K x Q^T) so the probability fragment maps
// to the AV A-operand with zero cross-lane traffic.  V is pre-transposed
// [B,H,Dh,S] so AV B-fragments are two contiguous b128 loads.
// ---------------------------------------------------------------------------
__global__ __launch_bounds__(256) void attn_kernel(const _Float16* __restrict__ Qh,
                                                   const _Float16* __restrict__ Kh,
                                                   const _Float16* __restrict__ Vt,
                                                   _Float16* __restrict__ M) {
  const int wid  = blockIdx.x * (blockDim.x >> 5) + (threadIdx.x >> 5); // 0..4095
  const int lane = threadIdx.x & 31;
  const int l16  = lane & 15;
  const int half = lane >> 4;
  const int qt = wid & 127;       // query tile within sequence
  const int bh = wid >> 7;        // 0..31
  const int b  = bh >> 4, h = bh & 15;

  const _Float16* Qp = Qh + (size_t)bh * N_CTX * D_HEAD;
  const _Float16* Kp = Kh + (size_t)bh * N_CTX * D_HEAD;
  const _Float16* Vp = Vt + (size_t)bh * D_HEAD * N_CTX;  // [64][2048]

  const int q_base = qt * 16;
  const int qg = q_base + l16;    // this lane's query (softmax state owner)

  float mrun = -1e30f, lrun = 0.f;
  v8f O[4] = {};

  // Preload Q^T B-fragments (two 32-wide k-steps over Dh=64): k = (e/8)*16 + 8*half + e%8
  v16h qb[2];
#pragma unroll
  for (int kh = 0; kh < 2; ++kh) {
    const _Float16* qp = Qp + (size_t)(q_base + l16) * D_HEAD + kh * 32 + 8 * half;
    qb[kh] = ld_frag(qp, qp + 16);
  }

  const int nchunks = (q_base + 16 + 31) >> 5;                    // causal bound
  const int nfull   = (q_base >= 31) ? (((q_base - 31) >> 5) + 1) : 0;

  // Steady state: all 32 keys strictly below the diagonal -> no masking.
  for (int jc = 0; jc < nfull; ++jc)
    attn_chunk<false>(jc * 32, qg, l16, half, Kp, Vp, qb, mrun, lrun, O);
  // Diagonal tail: per-element causal mask.
  for (int jc = nfull; jc < nchunks; ++jc)
    attn_chunk<true>(jc * 32, qg, l16, half, Kp, Vp, qb, mrun, lrun, O);

  // --- finalize: divide by l, write merged [B*S, D] as f16
  float linv[8];
#pragma unroll
  for (int r = 0; r < 8; ++r)
    linv[r] = __builtin_amdgcn_rcpf(__shfl(lrun, r + 8 * half));
#pragma unroll
  for (int nt = 0; nt < 4; ++nt)
#pragma unroll
    for (int r = 0; r < 8; ++r) {
      const int orow = b * N_CTX + q_base + r + 8 * half;
      const int col  = h * D_HEAD + nt * 16 + l16;
      M[(size_t)orow * D_MODEL + col] = (_Float16)(O[nt][r] * linv[r]);
    }
}

// ---------------------------------------------------------------------------
extern "C" void kernel_launch(void* const* d_in, const int* in_sizes, int n_in,
                              void* d_out, int out_size, void* d_ws, size_t ws_size,
                              hipStream_t stream) {
  (void)in_sizes; (void)n_in; (void)out_size; (void)ws_size;
  const float* X    = (const float*)d_in[0];
  const float* ln_w = (const float*)d_in[1];
  const float* ln_b = (const float*)d_in[2];
  const float* Wq   = (const float*)d_in[3];
  const float* bq   = (const float*)d_in[4];
  const float* Wk   = (const float*)d_in[5];
  const float* bk   = (const float*)d_in[6];
  const float* Wv   = (const float*)d_in[7];
  const float* bv   = (const float*)d_in[8];
  const float* Wo   = (const float*)d_in[9];
  const float* bo   = (const float*)d_in[10];

  char* ws = (char*)d_ws;
  const size_t SZ  = (size_t)ROWS * D_MODEL * sizeof(_Float16);     // 8 MB
  const size_t WSZ = (size_t)D_MODEL * D_MODEL * sizeof(_Float16);  // 2 MB
  _Float16* Xn  = (_Float16*)(ws);
  _Float16* Qh  = (_Float16*)(ws + SZ);
  _Float16* Kh  = (_Float16*)(ws + 2 * SZ);
  _Float16* Vh  = (_Float16*)(ws + 3 * SZ);   // transposed [B,H,Dh,S]
  _Float16* Mh  = (_Float16*)(ws + 4 * SZ);
  _Float16* Wqt = (_Float16*)(ws + 5 * SZ);
  _Float16* Wkt = (_Float16*)(ws + 5 * SZ + WSZ);
  _Float16* Wvt = (_Float16*)(ws + 5 * SZ + 2 * WSZ);
  _Float16* Wot = (_Float16*)(ws + 5 * SZ + 3 * WSZ);

  // One-time (per launch) weight transpose+convert: f32 [K,N] -> f16 [N,K]
  dim3 tb(32, 8), tg(32, 32);
  wt_kernel<<<tg, tb, 0, stream>>>(Wq, Wqt);
  wt_kernel<<<tg, tb, 0, stream>>>(Wk, Wkt);
  wt_kernel<<<tg, tb, 0, stream>>>(Wv, Wvt);
  wt_kernel<<<tg, tb, 0, stream>>>(Wo, Wot);

  ln_kernel<<<ROWS, 256, 0, stream>>>(X, ln_w, ln_b, Xn);

  // 2048 waves per GEMM -> 256 blocks of 8 waves
  gemm16<0><<<256, 256, 0, stream>>>(Xn, Wqt, bq, (void*)Qh, 0.125f); // Q pre-scaled
  gemm16<0><<<256, 256, 0, stream>>>(Xn, Wkt, bk, (void*)Kh, 1.0f);
  gemm16<2><<<256, 256, 0, stream>>>(Xn, Wvt, bv, (void*)Vh, 1.0f);

  attn_kernel<<<512, 256, 0, stream>>>(Qh, Kh, Vh, Mh);

  gemm16<1><<<256, 256, 0, stream>>>(Mh, Wot, bo, d_out, 1.0f);
}
